// Differential_Attention_17849884082840
// MI455X (gfx1250) — compile-verified, matrix operand order checked by reading, not run on hardware
//
#include <hip/hip_runtime.h>

typedef unsigned short u16;
typedef unsigned int   u32;
typedef unsigned long long u64;

typedef __attribute__((ext_vector_type(16))) __bf16 v16bf;
typedef __attribute__((ext_vector_type(8)))  float  v8f;
typedef __attribute__((ext_vector_type(4)))  unsigned int u32x4;
typedef __attribute__((ext_vector_type(8)))  int  i32x8;
typedef __attribute__((ext_vector_type(4)))  int  i32x4;

#define B_SZ      2
#define N_SEQ     2048
#define MODEL     1024
#define NHEAD     8
#define HD        128
#define DQK       64
#define PROJ      1024
#define QKV_COLS  3072
#define LAMBDA_INIT   0.78360576f   // 0.8 - 0.6*exp(-3.6)
#define ONE_MINUS_LI  0.21639424f
#define SCORE_SCALE   0.125f        // DQK^-0.5
#define EPS           1e-6f

#if __has_builtin(__builtin_amdgcn_tensor_load_to_lds)
#define USE_TDM 1
#else
#define USE_TDM 0
#endif

union BFrag { v16bf v; u16 u[16]; };

__device__ __forceinline__ u16 f2bf(float f) {
  u32 u = __float_as_uint(f);
  u32 r = (u + 0x7FFFu + ((u >> 16) & 1u)) >> 16;  // round-to-nearest-even
  return (u16)r;
}

// ISA 7.12.2: 16-bit A-matrix 16x32 per-lane K index for element e (0..15),
// half = lane>>4. Same map is used (mirrored M<->N) for 16-bit B 32x16.
__device__ __forceinline__ int kmap(int e, int half) {
  int j = e >> 1;
  return ((j & 4) ? 16 : 0) + (half ? 8 : 0) + ((j & 3) << 1) + (e & 1);
}

#if USE_TDM
// -------------------------------------------------- Tensor Data Mover -------
// 2-D tile DMA: global (h_rows x w_elems, row stride stride_elems, 2-byte
// elements) -> LDS (packed row-major). D# per ISA cdna5 8.3/8.4:
//   group0: count=1 | lds_addr | global_addr | type=2
//   group1: data_size=1(2B), tensor_dim0=w, tensor_dim1=h, tile_dim0=w,
//           tile_dim1=h, tensor_dim0_stride=stride, workgroup_mask=0
__device__ __forceinline__ void tdm_load_2d(u32 lds_addr, const void* gptr,
                                            u32 w, u32 h, u32 stride) {
  u64 ga = (u64)(size_t)gptr;
  u32x4 g0;
  g0[0] = 1u;                                            // count=1, user D#
  g0[1] = lds_addr;                                      // LDS byte address
  g0[2] = (u32)(ga & 0xFFFFFFFFu);                       // global_addr[31:0]
  g0[3] = (u32)((ga >> 32) & 0x01FFFFFFu) | (2u << 30);  // addr[56:32]|type=2
  i32x8 g1;
  g1[0] = (int)(1u << 16);                               // data_size=1 (2B)
  g1[1] = (int)((w & 0xFFFFu) << 16);                    // tensor_dim0 lo
  g1[2] = (int)((w >> 16) | ((h & 0xFFFFu) << 16));      // dim0 hi | dim1 lo
  g1[3] = (int)((h >> 16) | ((w & 0xFFFFu) << 16));      // dim1 hi | tile_dim0
  g1[4] = (int)(h & 0xFFFFu);                            // tile_dim1 (dim2=0)
  g1[5] = (int)stride;                                   // dim0_stride[31:0]
  g1[6] = 0;
  g1[7] = 0;
  i32x4 gz = {0, 0, 0, 0};
#if __clang_major__ >= 23
  i32x8 gz8 = {0, 0, 0, 0, 0, 0, 0, 0};
  __builtin_amdgcn_tensor_load_to_lds(g0, g1, gz, gz, gz8, 0);
#else
  __builtin_amdgcn_tensor_load_to_lds(g0, g1, gz, gz, 0);
#endif
}

__device__ __forceinline__ void tdm_wait0() {
#if __has_builtin(__builtin_amdgcn_s_wait_tensorcnt)
  __builtin_amdgcn_s_wait_tensorcnt(0);
#else
  asm volatile("s_wait_tensorcnt 0x0" ::: "memory");
#endif
}
__device__ __forceinline__ void tdm_wait2() {  // previous pair done; newest pair may fly
#if __has_builtin(__builtin_amdgcn_s_wait_tensorcnt)
  __builtin_amdgcn_s_wait_tensorcnt(2);
#else
  asm volatile("s_wait_tensorcnt 0x2" ::: "memory");
#endif
}

__device__ __forceinline__ u32 lds_off(const void* p) {
  // generic LDS address low 32 bits == LDS byte address (ISA 10.2 aperture map)
  return (u32)(size_t)p;
}
#endif  // USE_TDM

// ---------------------------------------------------------------- convert ---
__global__ __launch_bounds__(256)
void cvt_bf16_kernel(const float* __restrict__ in, u16* __restrict__ out, int n) {
  int i = blockIdx.x * 256 + threadIdx.x;
  if (i < n) out[i] = f2bf(in[i]);
}

// ------------------------------------------------------------------- GEMM ---
// C[M,Nc] = A[M,K](bf16) @ B[K,Nc](bf16); 128x128 block tile, 8 waves,
// wave w computes rows [16w,16w+16) x all 128 cols (8 accum frags).
// LDS tiles double-buffered; TDM pair for tile i+1 issued while tile i runs.
template <bool OUT_F32>
__global__ __launch_bounds__(256)
void gemm_bf16_kernel(const u16* __restrict__ A, const u16* __restrict__ Bm,
                      void* __restrict__ Cp, int M, int Nc, int K) {
  __shared__ __align__(16) u16 As[2][128 * 32];
  __shared__ __align__(16) u16 Bs[2][32 * 128];
  const int tid = threadIdx.x, wave = tid >> 5, lane = tid & 31;
  const int half = lane >> 4, l16 = lane & 15;
  const int brow = blockIdx.y * 128, bcol = blockIdx.x * 128;

  v8f vzero = {0.f, 0.f, 0.f, 0.f, 0.f, 0.f, 0.f, 0.f};
  v8f acc[8];
  for (int c = 0; c < 8; ++c) acc[c] = vzero;

  int cur = 0;
#if USE_TDM
  if (wave == 0) {  // prologue: fill buffer 0
    tdm_load_2d(lds_off(As[0]), A + (size_t)brow * K, 32, 128, (u32)K);
    tdm_load_2d(lds_off(Bs[0]), Bm + (size_t)bcol, 128, 32, (u32)Nc);
  }
#endif
  for (int k0 = 0; k0 < K; k0 += 32) {
    __syncthreads();  // everyone done reading buf[cur^1] from previous iter
#if USE_TDM
    if (wave == 0) {
      int kn = k0 + 32;
      if (kn < K) {
        tdm_load_2d(lds_off(As[cur ^ 1]), A + (size_t)brow * K + kn, 32, 128, (u32)K);
        tdm_load_2d(lds_off(Bs[cur ^ 1]), Bm + (size_t)kn * Nc + bcol, 128, 32, (u32)Nc);
        tdm_wait2();  // buf[cur] (issued last iter) complete; next pair in flight
      } else {
        tdm_wait0();
      }
    }
#else
    for (int q = tid; q < 512; q += 256) {
      int r = q >> 2, cs = q & 3;
      ((uint4*)As[cur])[q] = *(const uint4*)(A + (size_t)(brow + r) * K + k0 + cs * 8);
    }
    for (int q = tid; q < 512; q += 256) {
      int r = q >> 4, cs = q & 15;
      ((uint4*)Bs[cur])[q] = *(const uint4*)(Bm + (size_t)(k0 + r) * Nc + bcol + cs * 8);
    }
    if (k0 + 32 < K)
      __builtin_prefetch(A + (size_t)(brow + (tid >> 1)) * K + k0 + 32, 0, 1);
#endif
    __syncthreads();

    BFrag af;
    for (int e = 0; e < 16; ++e)
      af.u[e] = As[cur][(wave * 16 + l16) * 32 + kmap(e, half)];
    for (int c = 0; c < 8; ++c) {
      BFrag bfr;
      for (int e = 0; e < 16; ++e)
        bfr.u[e] = Bs[cur][kmap(e, half) * 128 + c * 16 + l16];
      acc[c] = __builtin_amdgcn_wmma_f32_16x16x32_bf16(
          false, af.v, false, bfr.v, (short)0, acc[c], false, false);
    }
    cur ^= 1;
  }

  // D layout: vgpr r -> M = r + 8*half, N = lane&15
  for (int c = 0; c < 8; ++c)
    for (int r = 0; r < 8; ++r) {
      int m = brow + wave * 16 + r + (half ? 8 : 0);
      int n = bcol + c * 16 + l16;
      float v = acc[c][r];
      if (OUT_F32) ((float*)Cp)[(size_t)m * Nc + n] = v;
      else         ((u16*)Cp)[(size_t)m * Nc + n] = f2bf(v);
    }
}

// -------------------------------------------------- differential attention ---
// grid: x = q-row block (N/128), y = head, z = batch. 8 waves; wave w owns
// 16 q rows. Flash-style online softmax, two streams, fused diff + RMSNorm.
// K/V tiles double-buffered via TDM, one key-block ahead of the WMMAs.
__global__ __launch_bounds__(256)
void diff_attn_kernel(const u16* __restrict__ qkv,
                      const float* __restrict__ lq1, const float* __restrict__ lk1,
                      const float* __restrict__ lq2, const float* __restrict__ lk2,
                      const float* __restrict__ rmsw, u16* __restrict__ outb) {
  __shared__ __align__(16) u16 kt[2][32 * HD];
  __shared__ __align__(16) u16 vt[2][32 * HD];
  __shared__ __align__(16) u16 pstage[8][16 * 32];

  const int tid = threadIdx.x, wave = tid >> 5, lane = tid & 31;
  const int half = lane >> 4, l16 = lane & 15;
  const int b = blockIdx.z, h = blockIdx.y;
  const int qb = blockIdx.x * 128 + wave * 16;
  const size_t rs = QKV_COLS;
  const u16* Qp = qkv + (size_t)b * N_SEQ * rs + (size_t)h * HD;
  const u16* Kp = Qp + PROJ;
  const u16* Vp = Qp + 2 * PROJ;

  // lambda (tiny redundant reduction per lane)
  float s1 = 0.f, s2 = 0.f;
  for (int i = 0; i < DQK; ++i) { s1 += lq1[i] * lk1[i]; s2 += lq2[i] * lk2[i]; }
  const float lam = __expf(s1) - __expf(s2) + LAMBDA_INIT;

  // Q fragments in registers: stream s (q1/q2), k-step t (d 0..31 / 32..63)
  BFrag qf[2][2];
  for (int s = 0; s < 2; ++s)
    for (int t = 0; t < 2; ++t)
      for (int e = 0; e < 16; ++e) {
        int d = s * DQK + t * 32 + kmap(e, half);
        qf[s][t].u[e] = Qp[(size_t)(qb + l16) * rs + d];
      }

  float m_[2][8], l_[2][8];
  v8f vzero = {0.f, 0.f, 0.f, 0.f, 0.f, 0.f, 0.f, 0.f};
  v8f o_[2][8];
  for (int s = 0; s < 2; ++s)
    for (int r = 0; r < 8; ++r) { m_[s][r] = -1e30f; l_[s][r] = 0.f; o_[s][r] = vzero; }

  int cur = 0;
#if USE_TDM
  if (wave == 0) {  // prologue: fill buffer 0 with key-block 0
    tdm_load_2d(lds_off(kt[0]), Kp, HD, 32, (u32)rs);
    tdm_load_2d(lds_off(vt[0]), Vp, HD, 32, (u32)rs);
  }
#endif
  for (int kb = 0; kb < N_SEQ; kb += 32) {
    __syncthreads();  // everyone done reading buf[cur^1] from previous iter
#if USE_TDM
    if (wave == 0) {
      int kn = kb + 32;
      if (kn < N_SEQ) {
        tdm_load_2d(lds_off(kt[cur ^ 1]), Kp + (size_t)kn * rs, HD, 32, (u32)rs);
        tdm_load_2d(lds_off(vt[cur ^ 1]), Vp + (size_t)kn * rs, HD, 32, (u32)rs);
        tdm_wait2();  // current buffer complete; next pair stays in flight
      } else {
        tdm_wait0();
      }
    }
#else
    for (int q = tid; q < 512; q += 256) {
      int r = q >> 4, cs = q & 15;
      ((uint4*)kt[cur])[q] = *(const uint4*)(Kp + (size_t)(kb + r) * rs + cs * 8);
      ((uint4*)vt[cur])[q] = *(const uint4*)(Vp + (size_t)(kb + r) * rs + cs * 8);
    }
#endif
    __syncthreads();

    for (int s = 0; s < 2; ++s) {
      // scores: 16 q rows x 32 keys (two 16-key subtiles), K=64 contraction
      v8f sc[2] = {vzero, vzero};
      for (int u = 0; u < 2; ++u)
        for (int t = 0; t < 2; ++t) {
          BFrag bfr;
          for (int e = 0; e < 16; ++e) {
            int d = s * DQK + t * 32 + kmap(e, half);
            bfr.u[e] = kt[cur][(u * 16 + l16) * HD + d];  // B[k=d][n=key]
          }
          sc[u] = __builtin_amdgcn_wmma_f32_16x16x32_bf16(
              false, qf[s][t].v, false, bfr.v, (short)0, sc[u], false, false);
        }

      // online softmax update for the 8 rows this lane-half owns
      for (int r = 0; r < 8; ++r) {
        float v0 = sc[0][r] * SCORE_SCALE, v1 = sc[1][r] * SCORE_SCALE;
        float bm = fmaxf(v0, v1);
        for (int msk = 1; msk < 16; msk <<= 1) bm = fmaxf(bm, __shfl_xor(bm, msk, 32));
        float mo = m_[s][r];
        float mn = fmaxf(mo, bm);
        float corr = __expf(mo - mn);
        float e0 = __expf(v0 - mn), e1 = __expf(v1 - mn);
        float rsum = e0 + e1;
        for (int msk = 1; msk < 16; msk <<= 1) rsum += __shfl_xor(rsum, msk, 32);
        l_[s][r] = l_[s][r] * corr + rsum;
        m_[s][r] = mn;
        for (int c = 0; c < 8; ++c) o_[s][c][r] *= corr;
        int prow = r + (half ? 8 : 0);
        pstage[wave][prow * 32 + l16]      = f2bf(e0);
        pstage[wave][prow * 32 + 16 + l16] = f2bf(e1);
      }
      // wave-local LDS RAW: drain DS counter before re-reading staged weights
      asm volatile("s_wait_dscnt 0" ::: "memory");

      // reshape D-layout weights -> A-layout frag, then attn @ V
      BFrag af;
      for (int e = 0; e < 16; ++e)
        af.u[e] = pstage[wave][l16 * 32 + kmap(e, half)];
      for (int c = 0; c < 8; ++c) {
        BFrag vfr;
        for (int e = 0; e < 16; ++e)
          vfr.u[e] = vt[cur][kmap(e, half) * HD + c * 16 + l16];
        o_[s][c] = __builtin_amdgcn_wmma_f32_16x16x32_bf16(
            false, af.v, false, vfr.v, (short)0, o_[s][c], false, false);
      }
    }
    cur ^= 1;
  }

  // epilogue: a1/l1 - lam*a2/l2, per-head RMSNorm, (1-lambda_init) scale
  float inv0[8], inv1[8];
  for (int r = 0; r < 8; ++r) { inv0[r] = 1.f / l_[0][r]; inv1[r] = lam / l_[1][r]; }
  float ss[8];
  for (int r = 0; r < 8; ++r) ss[r] = 0.f;
  for (int c = 0; c < 8; ++c)
    for (int r = 0; r < 8; ++r) {
      float v = o_[0][c][r] * inv0[r] - o_[1][c][r] * inv1[r];
      o_[0][c][r] = v;
      ss[r] += v * v;
    }
  for (int r = 0; r < 8; ++r)
    for (int msk = 1; msk < 16; msk <<= 1) ss[r] += __shfl_xor(ss[r], msk, 32);
  float rinv[8];
  for (int r = 0; r < 8; ++r) rinv[r] = rsqrtf(ss[r] * (1.f / HD) + EPS);

  for (int c = 0; c < 8; ++c) {
    float w = rmsw[c * 16 + l16] * ONE_MINUS_LI;
    for (int r = 0; r < 8; ++r) {
      int row = qb + r + (half ? 8 : 0);
      float fv = o_[0][c][r] * rinv[r] * w;
      outb[((size_t)b * N_SEQ + row) * PROJ + (size_t)h * HD + c * 16 + l16] = f2bf(fv);
    }
  }
}

// ------------------------------------------------------------------ launch ---
extern "C" void kernel_launch(void* const* d_in, const int* in_sizes, int n_in,
                              void* d_out, int out_size, void* d_ws, size_t ws_size,
                              hipStream_t stream) {
  (void)in_sizes; (void)n_in; (void)out_size; (void)ws_size;
  const float* x     = (const float*)d_in[0];
  const float* wqkv  = (const float*)d_in[1];
  const float* wproj = (const float*)d_in[2];
  const float* lq1   = (const float*)d_in[3];
  const float* lk1   = (const float*)d_in[4];
  const float* lq2   = (const float*)d_in[5];
  const float* lk2   = (const float*)d_in[6];
  const float* rmsw  = (const float*)d_in[7];
  float* out = (float*)d_out;

  const size_t NX   = (size_t)B_SZ * N_SEQ * MODEL;     // x elements
  const size_t NW1  = (size_t)MODEL * QKV_COLS;         // w_qkv
  const size_t NW2  = (size_t)PROJ * MODEL;             // w_proj
  const size_t NQKV = (size_t)B_SZ * N_SEQ * QKV_COLS;  // qkv

  u16* xb     = (u16*)d_ws;
  u16* wqkvb  = xb + NX;
  u16* wprojb = wqkvb + NW1;
  u16* qkvb   = wprojb + NW2;
  u16* aob    = qkvb + NQKV;

  cvt_bf16_kernel<<<(int)((NX  + 255) / 256), 256, 0, stream>>>(x,     xb,     (int)NX);
  cvt_bf16_kernel<<<(int)((NW1 + 255) / 256), 256, 0, stream>>>(wqkv,  wqkvb,  (int)NW1);
  cvt_bf16_kernel<<<(int)((NW2 + 255) / 256), 256, 0, stream>>>(wproj, wprojb, (int)NW2);

  // qkv = x @ w_qkv  (bf16 out)
  gemm_bf16_kernel<false><<<dim3(QKV_COLS / 128, (B_SZ * N_SEQ) / 128), 256, 0, stream>>>(
      xb, wqkvb, qkvb, B_SZ * N_SEQ, QKV_COLS, MODEL);

  // flash-style differential attention + RMSNorm (bf16 out)
  diff_attn_kernel<<<dim3(N_SEQ / 128, NHEAD, B_SZ), 256, 0, stream>>>(
      qkvb, lq1, lk1, lq2, lk2, rmsw, aob);

  // out = attn_out @ w_proj  (fp32 out)
  gemm_bf16_kernel<true><<<dim3(MODEL / 128, (B_SZ * N_SEQ) / 128), 256, 0, stream>>>(
      aob, wprojb, out, B_SZ * N_SEQ, MODEL, PROJ);
}
